// emgFeature_47004122087991
// MI455X (gfx1250) — matrix operand order repaired and checked by reading, not compiled, not run on hardware
//
#include <hip/hip_runtime.h>

// ---------------------------------------------------------------------------
// Loihi 2-layer SNN pipeline for MI455X (gfx1250, wave32, WMMA).
// Stages (stream-ordered):
//   k_prep  : W1,W2 f32 -> f16
//   k_gemm1 : a1[b][t][o] = W1 x spike            (v_wmma_f32_16x16x32_f16, K=16 zero-padded)
//   k_lif1  : LIF scan + axonal shift + frac delay -> s1d f16 [b][t][c]
//   k_gemm2 : a2[b][t][o] = W2 x s1d              (WMMA K=128; async global->LDS panel copy)
//   k_lif2  : LIF scan + delays, LDS-transposed store -> out f32 [b][c][t]
// Workspace: a_buf f32 64MB (reused for a1/a2) + s1d f16 32MB + weights (~36KB).
// ---------------------------------------------------------------------------

typedef __attribute__((ext_vector_type(16))) _Float16 v16h;
typedef __attribute__((ext_vector_type(8)))  _Float16 v8h;
typedef __attribute__((ext_vector_type(8)))  float    v8f;

static constexpr int   NB  = 64;     // batch
static constexpr int   T   = 2048;   // timesteps
static constexpr int   CIN = 16;     // layer-1 input channels
static constexpr int   CH  = 128;    // hidden/output channels
static constexpr int   TP  = 128;    // t-rows per GEMM block panel
static constexpr float AI  = 0.75f;     // current decay
static constexpr float AV  = 0.96875f;  // voltage decay

// ---------------------------------------------------------------- weights prep
__global__ void k_prep(const float* __restrict__ W1, const float* __restrict__ W2,
                       _Float16* __restrict__ W1h, _Float16* __restrict__ W2h) {
    int i = blockIdx.x * blockDim.x + threadIdx.x;
    if (i < CH * CIN) W1h[i] = (_Float16)W1[i];
    if (i < CH * CH)  W2h[i] = (_Float16)W2[i];
}

// ---------------------------------------------------------------- GEMM layer 1
// a1[b][t][o] = sum_c W1[o][c] * spike[b][c][t]    (M=t rows, N=o cols, K=c=16)
__global__ void __launch_bounds__(256) k_gemm1(const float* __restrict__ spike,
                                               const _Float16* __restrict__ W1h,
                                               float* __restrict__ a1) {
    constexpr int STR = 24;                 // halfs per LDS row (48B, 16B aligned)
    __shared__ _Float16 lA[TP * STR];

    const int blk = blockIdx.x;
    const int b   = blk >> 4;               // 16 panels of 128 t per batch elem
    const int t0  = (blk & 15) * TP;

    // Stage spike panel transposed to [tt][c] f16 (coalesced 512B global rows).
    for (int idx = threadIdx.x; idx < CIN * TP; idx += 256) {
        int c = idx >> 7, tt = idx & 127;
        lA[tt * STR + c] = (_Float16)spike[((size_t)(b * CIN + c)) * T + t0 + tt];
    }
    __syncthreads();

    const int lane = threadIdx.x & 31;
    const int sel  = lane >> 4;             // 0: lanes 0-15, 1: lanes 16-31
    const int l16  = lane & 15;
    const int o0   = (threadIdx.x >> 5) * 16;  // wave -> 16 output channels

    // B fragment = W1^T (32x16), K=16..31 is zero (lanes 16-31 all-zero).
    v16h bf = {};
    if (!sel) bf = *(const v16h*)(W1h + (o0 + l16) * CIN);   // contiguous 32B

    for (int ts = 0; ts < TP; ts += 16) {
        union { v16h v; v8h h[2]; } af;
        // A 16x32 f16, contiguous k order (same as B): lane<16 -> K{0..15}; lane>=16 -> K{16..31} (zero padding)
        af.h[0] = sel ? v8h{} : *(const v8h*)(lA + (ts + l16) * STR);
        af.h[1] = sel ? v8h{} : *(const v8h*)(lA + (ts + l16) * STR + 8);
        v8f c = {};
        c = __builtin_amdgcn_wmma_f32_16x16x32_f16(false, af.v, false, bf,
                                                   (short)0, c, false, false);
        // C layout: VGPR r, lanes0-15 -> (M=r, N=lane); lanes16-31 -> (M=r+8, N=lane-16)
        float* dst = a1 + ((size_t)(b * T + t0 + ts)) * CH + o0;
#pragma unroll
        for (int r = 0; r < 8; ++r)
            dst[(size_t)(r + sel * 8) * CH + l16] = c[r];
    }
}

// ---------------------------------------------------------------- GEMM layer 2
// a2[b][t][o] = sum_c W2[o][c] * s1d[b][t][c]      (K=128 -> 4 WMMA steps)
// A-panel (32KB) is staged via the CDNA5 async DMA path: per-lane
// global_load_async_to_lds_b128 (ASYNCcnt), no VGPR round-trip.
__global__ void __launch_bounds__(256) k_gemm2(const _Float16* __restrict__ s1d,
                                               const _Float16* __restrict__ W2h,
                                               float* __restrict__ a2) {
    constexpr int STRB = 272;               // bytes per LDS row (256 + 16 pad)
    __shared__ _Float16 lA[TP * (STRB / 2)];

    const int blk = blockIdx.x;
    const int b   = blk >> 4;
    const int t0  = (blk & 15) * TP;

    // Panel: s1d[b][t0..t0+127][0..127] f16 = contiguous 32KB, 16B per lane-op.
    // LDS operand = raw LDS byte offset = low 32 bits of the flat shared ptr
    // (aperture rule: the LDS address is the low 32 address bits).
    {
        const char*    gbase = (const char*)(s1d + ((size_t)(b * T + t0)) * CH);
        const uint32_t lbase = (uint32_t)(uintptr_t)(&lA[0]);
        for (int idx = threadIdx.x; idx < TP * 16; idx += 256) {  // 8 iters, no divergence
            int row = idx >> 4, part = idx & 15;
            uint64_t ga = (uint64_t)(uintptr_t)(gbase + row * 256 + part * 16);
            uint32_t la = lbase + (uint32_t)(row * STRB + part * 16);
            asm volatile("global_load_async_to_lds_b128 %0, %1, off"
                         :: "v"(la), "v"(ga) : "memory");
        }
        asm volatile("s_wait_asynccnt 0x0" ::: "memory");  // this wave's DMAs done
    }
    __syncthreads();                                       // cross-wave visibility

    const int lane = threadIdx.x & 31;
    const int sel  = lane >> 4;
    const int l16  = lane & 15;
    const int o0   = (threadIdx.x >> 5) * 16;

    // B fragments: W2^T, lane(l16) = column o0+l16, contiguous 32B per K-step half.
    v16h bf[4];
#pragma unroll
    for (int ks = 0; ks < 4; ++ks)
        bf[ks] = *(const v16h*)(W2h + (o0 + l16) * CH + ks * 32 + sel * 16);

    for (int ts = 0; ts < TP; ts += 16) {
        v8f c = {};
#pragma unroll
        for (int ks = 0; ks < 4; ++ks) {
            union { v16h v; v8h h[2]; } af;
            const char* base = (const char*)lA + (ts + l16) * STRB + ks * 64 + sel * 32;
            af.h[0] = *(const v8h*)(base);        // K  ks*32 + sel*16 .. +7
            af.h[1] = *(const v8h*)(base + 16);   // K  ks*32 + sel*16 + 8 .. +7
            c = __builtin_amdgcn_wmma_f32_16x16x32_f16(false, af.v, false, bf[ks],
                                                       (short)0, c, false, false);
        }
        float* dst = a2 + ((size_t)(b * T + t0 + ts)) * CH + o0;
#pragma unroll
        for (int r = 0; r < 8; ++r)
            dst[(size_t)(r + sel * 8) * CH + l16] = c[r];
    }
}

// ---------------------------------------------------------------- LIF layer 1
// One block per batch element; thread = channel. Reads a1[b][t][o] coalesced,
// writes delayed spikes s1d f16 [b][t][o] coalesced (256B per step).
__global__ void __launch_bounds__(128) k_lif1(const float* __restrict__ a1,
                                              const float* __restrict__ d1,
                                              _Float16* __restrict__ s1d) {
    const int b = blockIdx.x, o = threadIdx.x;
    const float dd   = d1[o];
    const int   fd   = (int)dd;              // floor, dd >= 0
    const float frac = dd - (float)fd;
    float w[6];
#pragma unroll
    for (int k = 0; k < 6; ++k)
        w[k] = (k == fd + 1) ? (1.f - frac) : ((k == fd + 2) ? frac : 0.f);

    float u = 0.f, v = 0.f, h1 = 0.f, h2 = 0.f, h3 = 0.f, h4 = 0.f, h5 = 0.f;
    const float* ap = a1 + (size_t)b * T * CH + o;
    _Float16*    sp = s1d + (size_t)b * T * CH + o;
    for (int t = 0; t < T; ++t) {
        float x = ap[(size_t)t * CH];
        u = AI * u + x;
        v = AV * v + u;
        float s = (v >= 1.f) ? 1.f : 0.f;
        v *= (1.f - s);
        // out[t] = (1-frac)*s[t-fd-1] + frac*s[t-fd-2], zero-filled history
        float outv = w[1] * h1 + w[2] * h2 + w[3] * h3 + w[4] * h4 + w[5] * h5;
        sp[(size_t)t * CH] = (_Float16)outv;
        h5 = h4; h4 = h3; h3 = h2; h2 = h1; h1 = s;
    }
}

// ---------------------------------------------------------------- LIF layer 2
// Same scan, but final output layout is [b][c][t]; stage 64-t tiles in padded
// LDS and store transposed so the 64MB write is coalesced.
__global__ void __launch_bounds__(128) k_lif2(const float* __restrict__ a2,
                                              const float* __restrict__ d2,
                                              float* __restrict__ out) {
    constexpr int TT = 64;
    __shared__ float tile[128 * (TT + 1)];   // +1 pad -> conflict-free writes

    const int b = blockIdx.x, o = threadIdx.x;
    const float dd   = d2[o];
    const int   fd   = (int)dd;
    const float frac = dd - (float)fd;
    float w[6];
#pragma unroll
    for (int k = 0; k < 6; ++k)
        w[k] = (k == fd + 1) ? (1.f - frac) : ((k == fd + 2) ? frac : 0.f);

    float u = 0.f, v = 0.f, h1 = 0.f, h2 = 0.f, h3 = 0.f, h4 = 0.f, h5 = 0.f;
    const float* ap = a2 + (size_t)b * T * CH + o;
    for (int t0 = 0; t0 < T; t0 += TT) {
        for (int j = 0; j < TT; ++j) {
            float x = ap[(size_t)(t0 + j) * CH];
            u = AI * u + x;
            v = AV * v + u;
            float s = (v >= 1.f) ? 1.f : 0.f;
            v *= (1.f - s);
            float outv = w[1] * h1 + w[2] * h2 + w[3] * h3 + w[4] * h4 + w[5] * h5;
            tile[o * (TT + 1) + j] = outv;
            h5 = h4; h4 = h3; h3 = h2; h2 = h1; h1 = s;
        }
        __syncthreads();
        for (int idx = threadIdx.x; idx < 128 * TT; idx += 128) {
            int oo = idx >> 6, j = idx & 63;
            out[((size_t)(b * CH + oo)) * T + t0 + j] = tile[oo * (TT + 1) + j];
        }
        __syncthreads();
    }
}

// ---------------------------------------------------------------- launcher
extern "C" void kernel_launch(void* const* d_in, const int* in_sizes, int n_in,
                              void* d_out, int out_size, void* d_ws, size_t ws_size,
                              hipStream_t stream) {
    const float* spike = (const float*)d_in[0];  // [64,16,2048]
    const float* W1    = (const float*)d_in[1];  // [128,16]
    const float* W2    = (const float*)d_in[2];  // [128,128]
    const float* d1    = (const float*)d_in[3];  // [128]
    const float* d2    = (const float*)d_in[4];  // [128]

    char* ws = (char*)d_ws;
    const size_t nA = (size_t)NB * T * CH;           // 16.7M elements
    float*    a_buf = (float*)ws;                    // 64 MB (reused a1 then a2)
    _Float16* s1d   = (_Float16*)(ws + nA * 4);      // 32 MB
    _Float16* W1h   = (_Float16*)(ws + nA * 4 + nA * 2);
    _Float16* W2h   = W1h + CH * CIN;
    (void)in_sizes; (void)n_in; (void)out_size; (void)ws_size;

    k_prep <<<64, 256, 0, stream>>>(W1, W2, W1h, W2h);
    k_gemm1<<<NB * (T / TP), 256, 0, stream>>>(spike, W1h, a_buf);
    k_lif1 <<<NB, 128, 0, stream>>>(a_buf, d1, s1d);
    k_gemm2<<<NB * (T / TP), 256, 0, stream>>>(s1d, W2h, a_buf);
    k_lif2 <<<NB, 128, 0, stream>>>(a_buf, d2, (float*)d_out);
}
